// PointNetSetAbstractionMsg_46119358825116
// MI455X (gfx1250) — matrix operand order repaired and checked by reading, not compiled, not run on hardware
//
#include <hip/hip_runtime.h>

typedef __attribute__((ext_vector_type(16))) _Float16 v16h;
typedef __attribute__((ext_vector_type(8)))  _Float16 v8h;
typedef __attribute__((ext_vector_type(8)))  float    v8f;

#define NPTS   4096
#define NCTR   1024
#define NBATCH 8
#define OUTCH  320

// ---------------------------------------------------------------------------
// Kernel 0 (prep, once per launch): pad+convert weights to f16 with WMMA-ready
// strides (cinPad+8 halves, 16B-aligned rows), fold BN into scale/shift.
// ---------------------------------------------------------------------------
__global__ __launch_bounds__(256) void prep_layer(const float* __restrict__ w,
                                                  const float* __restrict__ bi,
                                                  const float* __restrict__ g,
                                                  const float* __restrict__ be,
                                                  const float* __restrict__ rm,
                                                  const float* __restrict__ rv,
                                                  _Float16* __restrict__ wdst,
                                                  float* __restrict__ sc,
                                                  float* __restrict__ sh,
                                                  int cout, int cin, int cinPad)
{
    const int i = blockIdx.x * 256 + threadIdx.x;
    const int wst = cinPad + 8;
    if (i < cout * wst) {
        const int co = i / wst, ci = i - co * wst;
        const float v = (ci < cin) ? w[co * cin + ci] : 0.f;
        wdst[i] = (_Float16)v;
    }
    if (i < cout) {
        const float s = g[i] * rsqrtf(rv[i] + 1e-5f);
        sc[i] = s;
        sh[i] = (bi[i] - rm[i]) * s + be[i];
    }
}

// ---------------------------------------------------------------------------
// Kernel 1: farthest point sampling. One workgroup per batch, points in LDS.
// Emits new_xyz (ws, [B,S,3]) and the first output tensor ([B,3,S]).
// ---------------------------------------------------------------------------
__global__ __launch_bounds__(256) void fps_kernel(const float* __restrict__ pc,
                                                  float* __restrict__ nxyz,
                                                  float* __restrict__ out)
{
    __shared__ float lx[NPTS], ly[NPTS], lz[NPTS];
    __shared__ float rd[256];
    __shared__ int   ri[256];
    __shared__ int   curf;

    const int b = blockIdx.x, tid = threadIdx.x;
    const float* pb = pc + (size_t)b * 3 * NPTS;
    for (int n = tid; n < NPTS; n += 256) {
        lx[n] = pb[n]; ly[n] = pb[NPTS + n]; lz[n] = pb[2 * NPTS + n];
    }
    if (tid == 0) curf = 0;
    __syncthreads();

    float dist[16];
#pragma unroll
    for (int j = 0; j < 16; ++j) dist[j] = 1e10f;
    const int base = tid * 16;

    for (int it = 0; it < NCTR; ++it) {
        const int f = curf;
        const float cx = lx[f], cy = ly[f], cz = lz[f];
        if (tid == 0) {
            nxyz[((size_t)b * NCTR + it) * 3 + 0] = cx;
            nxyz[((size_t)b * NCTR + it) * 3 + 1] = cy;
            nxyz[((size_t)b * NCTR + it) * 3 + 2] = cz;
            out[(size_t)b * 3 * NCTR + 0 * NCTR + it] = cx;
            out[(size_t)b * 3 * NCTR + 1 * NCTR + it] = cy;
            out[(size_t)b * 3 * NCTR + 2 * NCTR + it] = cz;
        }
        float bd = -1.f; int bi2 = 0;
#pragma unroll
        for (int j = 0; j < 16; ++j) {
            const int n = base + j;
            const float dx = lx[n] - cx, dy = ly[n] - cy, dz = lz[n] - cz;
            const float d = dx * dx + dy * dy + dz * dz;
            dist[j] = fminf(dist[j], d);
            if (dist[j] > bd) { bd = dist[j]; bi2 = n; }   // ascending n => first max
        }
        rd[tid] = bd; ri[tid] = bi2;
        __syncthreads();
        for (int st = 128; st > 0; st >>= 1) {
            if (tid < st) {
                const float d2 = rd[tid + st]; const int i2 = ri[tid + st];
                if (d2 > rd[tid] || (d2 == rd[tid] && i2 < ri[tid])) { rd[tid] = d2; ri[tid] = i2; }
            }
            __syncthreads();
        }
        if (tid == 0) curf = ri[0];
        __syncthreads();
    }
}

// Load a WMMA fragment as two aligned 16B LDS vectors (=> ds_load_b128).
__device__ __forceinline__ v16h ld2(const _Float16* p, int step8)
{
    const v8h* q = (const v8h*)p;
    const v8h x = q[0], y = q[step8];
    return __builtin_shufflevector(x, y, 0, 1, 2, 3, 4, 5, 6, 7,
                                         8, 9, 10, 11, 12, 13, 14, 15);
}

// ---------------------------------------------------------------------------
// WMMA MLP layer: rows(K samples) x cout, reduction over kChunks*32 channels.
// Fragments per ISA 7.12.2 layouts; folded BN (scale/shift) + ReLU on D.
// Uniform per-wave tile loop: EXEC stays all-ones around v_wmma.
// ---------------------------------------------------------------------------
__device__ __forceinline__ void mlp_layer(const _Float16* __restrict__ in, int inStride,
                                          _Float16* __restrict__ outp, int outStride,
                                          const _Float16* __restrict__ w, int wStride,
                                          int rows, int cout, int kChunks,
                                          const float* __restrict__ sc,
                                          const float* __restrict__ sh, int tid)
{
    const int lane = tid & 31, wave = tid >> 5;
    const int mTiles = rows >> 4, nTiles = cout >> 4;
    const bool lo = lane < 16;
    const int lr = lane & 15;

    for (int t = wave; t < mTiles * nTiles; t += 8) {
        const int mt = t / nTiles, nt = t - mt * nTiles;
        v8f acc = {};
        const _Float16* arow = in + (mt * 16 + lr) * inStride + (lo ? 0 : 8);
        const _Float16* brow = w + (nt * 16 + lr) * wStride + (lo ? 0 : 8);
        for (int kc = 0; kc < kChunks; ++kc) {
            const v16h a  = ld2(arow + kc * 32, 2);   // A 16x32: +0 and +16 halves
            const v16h bb = ld2(brow + kc * 32, 2);   // B 32x16: +0 and +16 halves
            acc = __builtin_amdgcn_wmma_f32_16x16x32_f16(false, a, false, bb,
                                                         (short)0, acc, false, false);
        }
        const int c = nt * 16 + lr;
        const float s = sc[c], h = sh[c];
        _Float16* orow = outp + (mt * 16 + (lo ? 0 : 8)) * outStride + c;
#pragma unroll
        for (int r = 0; r < 8; ++r) {                 // f32 C/D: vgpr r -> M=r / M=r+8
            const float v = acc[r] * s + h;
            orow[r * outStride] = (_Float16)(v > 0.f ? v : 0.f);
        }
    }
}

struct ScaleW {
    const _Float16* w;                       // contiguous 3-layer padded f16 blob
    const float *sc0, *sh0, *sc1, *sh1, *sc2, *sh2;
};

// ---------------------------------------------------------------------------
// Kernel 2 (x3 scales): async weight stage (overlapped with ball query/gather)
// + fused ball-query + gather + 3-layer WMMA MLP + maxpool.
// One workgroup (256 thr = 8 waves) per (b, s). Dynamic LDS (~126 KB max).
// ---------------------------------------------------------------------------
template <int K, int C1, int C2, int C3, int COFF>
__global__ __launch_bounds__(256) void group_mlp_kernel(const float* __restrict__ pc,
                                                        const float* __restrict__ ft,
                                                        const float* __restrict__ nxyz,
                                                        ScaleW S, float r2,
                                                        float* __restrict__ out)
{
    constexpr int XST = 40, BST = 136;
    constexpr int W0H = C1 * 40;             // layer0 blob (stride 32+8)
    constexpr int W1H = C2 * (C1 + 8);       // layer1 blob
    constexpr int W2H = C3 * (C2 + 8);       // layer2 blob
    constexpr int WBYTES = (W0H + W1H + W2H) * 2;

    extern __shared__ char smem[];
    _Float16* xin  = (_Float16*)smem;            // [K][XST] input rows, 9ch -> 32
    _Float16* bufA = xin  + K * XST;             // [K][BST]
    _Float16* bufB = bufA + K * BST;             // [K][BST]
    _Float16* wlds = bufB + K * BST;             // padded f16 weights, all 3 layers
    int*      gidx = (int*)(wlds + W0H + W1H + W2H);
    float*    cent = (float*)(gidx + K);

    const int b = blockIdx.x >> 10, s = blockIdx.x & 1023;
    const int tid = threadIdx.x, lane = tid & 31, wave = tid >> 5;

    // ---- kick off async weight staging: global -> LDS, tracked by ASYNCcnt ----
    {
        const unsigned long long gbase = (unsigned long long)(const void*)S.w;
        const unsigned ldsbase = (unsigned)(size_t)(void*)wlds;
        for (int off = tid * 16; off < WBYTES; off += 256 * 16) {
            asm volatile("global_load_async_to_lds_b128 %0, %1, %2"
                         :: "v"(ldsbase + (unsigned)off), "v"(off), "s"(gbase)
                         : "memory");
        }
    }

    if (tid < 3) cent[tid] = nxyz[((size_t)b * NCTR + s) * 3 + tid];
    __syncthreads();

    // ---- ball query: first K in-range indices in ascending order (wave 0) ----
    if (wave == 0) {
        const float cx = cent[0], cy = cent[1], cz = cent[2];
        const float* pb = pc + (size_t)b * 3 * NPTS;
        int cnt = 0;
        for (int bs = 0; bs < NPTS && cnt < K; bs += 32) {
            const int n = bs + lane;
            const float dx = pb[n] - cx, dy = pb[NPTS + n] - cy, dz = pb[2 * NPTS + n] - cz;
            const bool inr = (dx * dx + dy * dy + dz * dz) <= r2;
            const unsigned m = __builtin_amdgcn_ballot_w32(inr);
            const int pre = __popc(m & ((1u << lane) - 1u));
            if (inr && (cnt + pre) < K) gidx[cnt + pre] = n;
            cnt += __popc(m);
        }
        const int cc = cnt < K ? cnt : K;
        const int first = (cc > 0) ? gidx[0] : (NPTS - 1);
        for (int j = lane; j < K; j += 32)
            if (j >= cc || gidx[j] == NPTS - 1) gidx[j] = first;
    }
    __syncthreads();

    // ---- gather: 6 features + 3 relative coords, pad to 32 channels (f16) ----
    if (tid < K) {
        const int g = gidx[tid];
        const float* fb = ft + (size_t)b * 6 * NPTS;
        const float* pb = pc + (size_t)b * 3 * NPTS;
        _Float16* xr = xin + tid * XST;
#pragma unroll
        for (int c = 0; c < 6; ++c) xr[c] = (_Float16)fb[c * NPTS + g];
#pragma unroll
        for (int c = 0; c < 3; ++c) xr[6 + c] = (_Float16)(pb[c * NPTS + g] - cent[c]);
#pragma unroll
        for (int c = 9; c < 32; ++c) xr[c] = (_Float16)0.f;
    }

    // ---- weights must be resident before first WMMA ----
    asm volatile("s_wait_asynccnt 0" ::: "memory");
    __syncthreads();

    const _Float16* wl0 = wlds;
    const _Float16* wl1 = wlds + W0H;
    const _Float16* wl2 = wlds + W0H + W1H;

    mlp_layer(xin, XST, bufA, BST, wl0, 40, K, C1, 1, S.sc0, S.sh0, tid);
    __syncthreads();
    mlp_layer(bufA, BST, bufB, BST, wl1, C1 + 8, K, C2, C1 / 32, S.sc1, S.sh1, tid);
    __syncthreads();
    mlp_layer(bufB, BST, bufA, BST, wl2, C2 + 8, K, C3, C2 / 32, S.sc2, S.sh2, tid);
    __syncthreads();

    // ---- max over K samples, write new_feat channel block ----
    if (tid < C3) {
        float m = (float)bufA[tid];
        for (int k2 = 1; k2 < K; ++k2) m = fmaxf(m, (float)bufA[k2 * BST + tid]);
        out[(size_t)NBATCH * 3 * NCTR + ((size_t)b * OUTCH + COFF + tid) * NCTR + s] = m;
    }
}

static constexpr size_t grp_smem_bytes(int K, int C1, int C2, int C3)
{
    const int WH = C1 * 40 + C2 * (C1 + 8) + C3 * (C2 + 8);
    return (size_t)(K * 40 + 2 * K * 136 + WH) * 2 + (size_t)K * 4 + 16;
}

extern "C" void kernel_launch(void* const* d_in, const int* in_sizes, int n_in,
                              void* d_out, int out_size, void* d_ws, size_t ws_size,
                              hipStream_t stream)
{
    (void)in_sizes; (void)n_in; (void)out_size; (void)ws_size;
    const float* pc = (const float*)d_in[0];
    const float* ft = (const float*)d_in[1];
    float* out = (float*)d_out;

    // ---- workspace layout ----
    char* wsb = (char*)d_ws;
    float*     nxyz  = (float*)wsb;                       // [B,S,3] f32   (96 KB)
    _Float16*  wblob = (_Float16*)(wsb + 98304);          // padded f16 weights (~87 KB)
    float*     scsh  = (float*)(wsb + 98304 + 90112);     // folded BN scale/shift

    static const int COUT[3][3] = {{32, 32, 64}, {64, 64, 128}, {64, 96, 128}};
    static const int CIN [3][3] = {{ 9, 32, 32}, { 9, 64,  64}, { 9, 64,  96}};
    static const int CPAD[3][3] = {{32, 32, 32}, {32, 64,  64}, {32, 64,  96}};

    _Float16* wsc[3];       // per-scale weight blob base
    const float* sp[3][3][2];
    {
        size_t woff = 0, soff = 0;
        for (int i = 0; i < 3; ++i) {
            wsc[i] = wblob + woff;
            for (int j = 0; j < 3; ++j) {
                const int base = 2 + (i * 3 + j) * 6;     // dict order: w,b,g,beta,rm,rv
                const float* w  = (const float*)d_in[base + 0];
                const float* bi = (const float*)d_in[base + 1];
                const float* g  = (const float*)d_in[base + 2];
                const float* be = (const float*)d_in[base + 3];
                const float* rm = (const float*)d_in[base + 4];
                const float* rv = (const float*)d_in[base + 5];
                _Float16* wdst = wblob + woff;
                float* sc = scsh + soff; soff += COUT[i][j];
                float* sh = scsh + soff; soff += COUT[i][j];
                sp[i][j][0] = sc; sp[i][j][1] = sh;
                const int elems = COUT[i][j] * (CPAD[i][j] + 8);
                woff += elems;
                prep_layer<<<(elems + 255) / 256, 256, 0, stream>>>(
                    w, bi, g, be, rm, rv, wdst, sc, sh, COUT[i][j], CIN[i][j], CPAD[i][j]);
            }
        }
    }

    fps_kernel<<<NBATCH, 256, 0, stream>>>(pc, nxyz, out);

    ScaleW S0{wsc[0], sp[0][0][0], sp[0][0][1], sp[0][1][0], sp[0][1][1], sp[0][2][0], sp[0][2][1]};
    ScaleW S1{wsc[1], sp[1][0][0], sp[1][0][1], sp[1][1][0], sp[1][1][1], sp[1][2][0], sp[1][2][1]};
    ScaleW S2{wsc[2], sp[2][0][0], sp[2][0][1], sp[2][1][0], sp[2][1][1], sp[2][2][0], sp[2][2][1]};

    group_mlp_kernel<16, 32, 32, 64, 0>
        <<<NBATCH * NCTR, 256, grp_smem_bytes(16, 32, 32, 64), stream>>>(pc, ft, nxyz, S0, 0.25f, out);
    group_mlp_kernel<32, 64, 64, 128, 64>
        <<<NBATCH * NCTR, 256, grp_smem_bytes(32, 64, 64, 128), stream>>>(pc, ft, nxyz, S1, 1.0f, out);
    group_mlp_kernel<128, 64, 96, 128, 192>
        <<<NBATCH * NCTR, 256, grp_smem_bytes(128, 64, 96, 128), stream>>>(pc, ft, nxyz, S2, 4.0f, out);
}
